// LocalEncoder_13872744366805
// MI455X (gfx1250) — matrix operand order repaired and checked by reading, not compiled, not run on hardware
//
#include <hip/hip_runtime.h>
#include <hip/hip_bf16.h>

// ---------------------------------------------------------------------------
// LocalEncoder on MI455X (gfx1250): f16 WMMA GEMMs (async-LDS double-buffered)
// + flash-style windowed attention.
// ---------------------------------------------------------------------------

typedef __attribute__((ext_vector_type(16))) _Float16 v16h;
typedef __attribute__((ext_vector_type(8)))  _Float16 v8h;
typedef __attribute__((ext_vector_type(8)))  float    v8f;

#define H_DIM   512
#define NHEAD   8
#define HD      64
#define SEQ     2048
#define BATCH   4
#define WINSZ   512
#define NVOCAB  30000
#define NPATCH  8
#define LPATCH  (SEQ / NPATCH)     // 256
#define ROWS    (BATCH * SEQ)      // 8192

// Build a 16x(f16) fragment from two contiguous 8x(f16) chunks.
__device__ __forceinline__ v16h ldfrag(const _Float16* p0, const _Float16* p1) {
    v8h a = *(const v8h*)p0;
    v8h b = *(const v8h*)p1;
    v16h r;
#pragma unroll
    for (int i = 0; i < 8; ++i) { r[i] = a[i]; r[i + 8] = b[i]; }
    return r;
}

__device__ __forceinline__ v8f wmma_f16(v16h a, v16h b, v8f c) {
    return __builtin_amdgcn_wmma_f32_16x16x32_f16(false, a, false, b, (short)0, c,
                                                  false, false);
}

// Async 16B copy global -> LDS (GLOBAL_LOAD_ASYNC_TO_LDS_B128, ASYNCcnt).
// LDS dest VGPR carries the workgroup-relative LDS byte offset = low 32 bits
// of the flat address of a __shared__ object (LDS aperture lives in addr[63:32]).
__device__ __forceinline__ void async_cp16(void* dst_lds, const void* src_global) {
    unsigned off = (unsigned)(uintptr_t)dst_lds;
    asm volatile("global_load_async_to_lds_b128 %0, %1, off"
                 :: "v"(off), "v"(src_global) : "memory");
}
__device__ __forceinline__ void async_wait0() {
    asm volatile("s_wait_asynccnt 0x0" ::: "memory");
}

// ---------------------------------------------------------------------------
// f32 -> f16 cast
// ---------------------------------------------------------------------------
__global__ void cast_h_kernel(const float* __restrict__ s, _Float16* __restrict__ d, int n) {
    int i = blockIdx.x * 256 + threadIdx.x;
    if (i < n) d[i] = (_Float16)s[i];
}

// ---------------------------------------------------------------------------
// Embedding: byte_emb gather + rolling-hash ngram gathers (n = 3,4,5)
// ---------------------------------------------------------------------------
__global__ __launch_bounds__(128)
void embed_kernel(const int* __restrict__ seq, const float* __restrict__ be,
                  const float* __restrict__ ng3, const float* __restrict__ ng4,
                  const float* __restrict__ ng5,
                  float* __restrict__ x, _Float16* __restrict__ xh) {
    int tok = blockIdx.x;
    int b = tok >> 11;
    int s = tok & (SEQ - 1);
    __shared__ int sh3, sh4, sh5, sby;
    if (threadIdx.x == 0) {
        const int* p = seq + (size_t)b * SEQ;
        sby = p[s];
        unsigned int acc = 0u, pw = 1u;
        int h3 = -1, h4 = -1, h5 = -1;
#pragma unroll
        for (int i = 0; i < 5; ++i) {
            if (s + i < SEQ) acc += (unsigned int)p[s + i] * pw;
            pw *= 256u;  // wraps: 256^4 mod 2^32 == 0, matching the reference
            if (i == 2 && s + 2 < SEQ) h3 = (int)(acc % (unsigned)NVOCAB);
            if (i == 3 && s + 3 < SEQ) h4 = (int)(acc % (unsigned)NVOCAB);
            if (i == 4 && s + 4 < SEQ) h5 = (int)(acc % (unsigned)NVOCAB);
        }
        sh3 = h3; sh4 = h4; sh5 = h5;
    }
    __syncthreads();
    int c = threadIdx.x * 4;
    float4 v = *(const float4*)(be + (size_t)sby * H_DIM + c);
    float vr[4] = {v.x, v.y, v.z, v.w};
    const float sc = 0.25f;  // 1/(len(NGRAMS)+1)
    if (sh3 >= 0) {
        float4 g = *(const float4*)(ng3 + (size_t)sh3 * H_DIM + c);
        vr[0] += sc * g.x; vr[1] += sc * g.y; vr[2] += sc * g.z; vr[3] += sc * g.w;
    }
    if (sh4 >= 0) {
        float4 g = *(const float4*)(ng4 + (size_t)sh4 * H_DIM + c);
        vr[0] += sc * g.x; vr[1] += sc * g.y; vr[2] += sc * g.z; vr[3] += sc * g.w;
    }
    if (sh5 >= 0) {
        float4 g = *(const float4*)(ng5 + (size_t)sh5 * H_DIM + c);
        vr[0] += sc * g.x; vr[1] += sc * g.y; vr[2] += sc * g.z; vr[3] += sc * g.w;
    }
    size_t o = (size_t)tok * H_DIM + c;
#pragma unroll
    for (int k = 0; k < 4; ++k) {
        x[o + k]  = vr[k];
        xh[o + k] = (_Float16)vr[k];
    }
}

// ---------------------------------------------------------------------------
// WMMA GEMM: out[M,N] = A[M,K] * W[N,K]^T + bias, optional ReLU.
// Workgroup tile 128x64, 8 waves (4 along M x 2 along N), each wave 32x32 of C
// (4 WMMAs / K-step). K-steps of 32. Double-buffered LDS filled with
// global_load_async_to_lds_b128 so the next tile's DMA overlaps compute.
// ---------------------------------------------------------------------------
__global__ __launch_bounds__(256)
void gemm_wmma_kernel(const _Float16* __restrict__ A, const _Float16* __restrict__ Wm,
                      const float* __restrict__ bias,
                      float* __restrict__ outF, _Float16* __restrict__ outH,
                      int M, int N, int K, int relu) {
    __shared__ __align__(16) _Float16 lA[2][128 * 32];
    __shared__ __align__(16) _Float16 lB[2][64 * 32];
    int n0 = blockIdx.x * 64;
    int m0 = blockIdx.y * 128;
    int t = threadIdx.x;
    int lane = t & 31, wv = t >> 5;
    int hi = lane >> 4, lo16 = lane & 15;
    int mw = (wv & 3) * 32;        // wave's M offset in tile (0,32,64,96)
    int nw = (wv >> 2) * 32;       // wave's N offset in tile (0,32)

    v8f acc00, acc01, acc10, acc11;
#pragma unroll
    for (int e = 0; e < 8; ++e) {
        acc00[e] = 0.0f; acc01[e] = 0.0f; acc10[e] = 0.0f; acc11[e] = 0.0f;
    }

    // Per-thread staging chunks (16B each): 2 for the 128x32 A tile, 1 for B.
    int ra0 = t >> 2,          ca0 = (t & 3) * 8;           // A rows 0..63
    int ra1 = (t + 256) >> 2,  ca1 = ca0;                   // A rows 64..127
    int rb  = t >> 2,          cb  = ca0;                   // B rows 0..63

    auto stage = [&](int kt, int buf) {
        int kc = kt * 32;
        async_cp16(&lA[buf][ra0 * 32 + ca0], &A [(size_t)(m0 + ra0) * K + kc + ca0]);
        async_cp16(&lA[buf][ra1 * 32 + ca1], &A [(size_t)(m0 + ra1) * K + kc + ca1]);
        async_cp16(&lB[buf][rb  * 32 + cb ], &Wm[(size_t)(n0 + rb ) * K + kc + cb ]);
    };

    int ksteps = K >> 5;
    stage(0, 0);
    async_wait0();
    __syncthreads();

    for (int kt = 0; kt < ksteps; ++kt) {
        int buf = kt & 1;
        if (kt + 1 < ksteps) stage(kt + 1, buf ^ 1);   // overlap DMA with WMMA

        const _Float16* ar0 = &lA[buf][(mw + lo16) * 32];
        const _Float16* ar1 = &lA[buf][(mw + 16 + lo16) * 32];
        v16h af0 = ldfrag(ar0 + 8 * hi, ar0 + 16 + 8 * hi);
        v16h af1 = ldfrag(ar1 + 8 * hi, ar1 + 16 + 8 * hi);
        const _Float16* br0 = &lB[buf][(nw + lo16) * 32];
        const _Float16* br1 = &lB[buf][(nw + 16 + lo16) * 32];
        v16h bf0 = ldfrag(br0 + 8 * hi, br0 + 16 + 8 * hi);
        v16h bf1 = ldfrag(br1 + 8 * hi, br1 + 16 + 8 * hi);

        acc00 = wmma_f16(af0, bf0, acc00);
        acc01 = wmma_f16(af0, bf1, acc01);
        acc10 = wmma_f16(af1, bf0, acc10);
        acc11 = wmma_f16(af1, bf1, acc11);

        async_wait0();      // next tile's DMA has landed in LDS
        __syncthreads();
    }

    int col0 = n0 + nw + lo16;
    int col1 = col0 + 16;
    float b0 = bias[col0], b1 = bias[col1];
#pragma unroll
    for (int mi = 0; mi < 2; ++mi) {
        v8f aN0 = mi ? acc10 : acc00;
        v8f aN1 = mi ? acc11 : acc01;
#pragma unroll
        for (int e = 0; e < 8; ++e) {
            int row = m0 + mw + mi * 16 + e + 8 * hi;
            float f0 = aN0[e] + b0;
            float f1 = aN1[e] + b1;
            if (relu) { f0 = fmaxf(f0, 0.0f); f1 = fmaxf(f1, 0.0f); }
            size_t o0 = (size_t)row * N + col0;
            size_t o1 = (size_t)row * N + col1;
            if (outF) { outF[o0] = f0; outF[o1] = f1; }
            if (outH) { outH[o0] = (_Float16)f0; outH[o1] = (_Float16)f1; }
        }
    }
}

// ---------------------------------------------------------------------------
// Windowed causal attention, flash-style, one wave per (b, h, 16-query tile).
// qkvh: (B*S, 3H) f16 interleaved [q|k|v]. Output ah: (B*S, H) f16.
// ---------------------------------------------------------------------------
__global__ __launch_bounds__(32)
void attn_kernel(const _Float16* __restrict__ qkvh, _Float16* __restrict__ ah) {
    __shared__ __align__(16) _Float16 lds_p[16 * 32];    // P tile (C->A relayout)
    __shared__ __align__(16) _Float16 lds_vt[64 * 32];   // V^T block

    int bid = blockIdx.x;
    int qt = bid & (SEQ / 16 - 1);
    int h  = (bid >> 7) & (NHEAD - 1);
    int b  = bid >> 10;
    int q0 = qt * 16;

    int lane = threadIdx.x;
    int hi = lane >> 4, lo16 = lane & 15;

    // Q fragments: 2 K-steps over hd (0..31, 32..63)
    const _Float16* qrow = qkvh + ((size_t)(b * SEQ + q0 + lo16)) * (3 * H_DIM) + h * HD;
    v16h qf0 = ldfrag(qrow + 8 * hi,      qrow + 16 + 8 * hi);
    v16h qf1 = ldfrag(qrow + 32 + 8 * hi, qrow + 48 + 8 * hi);

    v8f accv[4];
#pragma unroll
    for (int tI = 0; tI < 4; ++tI)
#pragma unroll
        for (int e = 0; e < 8; ++e) accv[tI][e] = 0.0f;
    float mrow[8], lrow[8];
#pragma unroll
    for (int e = 0; e < 8; ++e) { mrow[e] = -1e30f; lrow[e] = 0.0f; }

    int lo = q0 - WINSZ;
    int kb0 = lo > 0 ? (lo & ~31) : 0;
    for (int kb = kb0; kb <= q0 + 15; kb += 32) {
        // stage V^T block: lds_vt[hd][kk], kk in [0,32)
        for (int idx = lane; idx < 32 * 64; idx += 32) {
            int kk = idx >> 6, hd = idx & 63;
            lds_vt[hd * 32 + kk] =
                qkvh[((size_t)(b * SEQ + kb + kk)) * (3 * H_DIM) + 2 * H_DIM + h * HD + hd];
        }

        // scores: two 16-key subtiles, each 2 WMMAs over hd
        float p0e[8], p1e[8];
        float mblk[8];
#pragma unroll
        for (int e = 0; e < 8; ++e) mblk[e] = -1e30f;
        v8f s0; v8f s1;
        {
            const _Float16* kr = qkvh + ((size_t)(b * SEQ + kb + lo16)) * (3 * H_DIM)
                                 + H_DIM + h * HD;
            v16h kf0 = ldfrag(kr + 8 * hi,      kr + 16 + 8 * hi);
            v16h kf1 = ldfrag(kr + 32 + 8 * hi, kr + 48 + 8 * hi);
            v8f a;
#pragma unroll
            for (int e = 0; e < 8; ++e) a[e] = 0.0f;
            a = wmma_f16(qf0, kf0, a);
            a = wmma_f16(qf1, kf1, a);
            s0 = a;
        }
        {
            const _Float16* kr = qkvh + ((size_t)(b * SEQ + kb + 16 + lo16)) * (3 * H_DIM)
                                 + H_DIM + h * HD;
            v16h kf0 = ldfrag(kr + 8 * hi,      kr + 16 + 8 * hi);
            v16h kf1 = ldfrag(kr + 32 + 8 * hi, kr + 48 + 8 * hi);
            v8f a;
#pragma unroll
            for (int e = 0; e < 8; ++e) a[e] = 0.0f;
            a = wmma_f16(qf0, kf0, a);
            a = wmma_f16(qf1, kf1, a);
            s1 = a;
        }

        // mask + row max (rows m = e + 8*hi; cols n = lane&15 (+16))
        int msk0[8], msk1[8];
#pragma unroll
        for (int e = 0; e < 8; ++e) {
            int i = q0 + e + 8 * hi;
            int j0 = kb + lo16;
            int j1 = kb + 16 + lo16;
            float v0 = s0[e] * 0.125f;
            float v1 = s1[e] * 0.125f;
            msk0[e] = (j0 <= i) && (j0 >= i - WINSZ);
            msk1[e] = (j1 <= i) && (j1 >= i - WINSZ);
            v0 = msk0[e] ? v0 : -1e30f;
            v1 = msk1[e] ? v1 : -1e30f;
            s0[e] = v0; s1[e] = v1;
            float rv = fmaxf(v0, v1);
#pragma unroll
            for (int off = 1; off < 16; off <<= 1)
                rv = fmaxf(rv, __shfl_xor(rv, off, 16));
            mblk[e] = rv;
        }

        // online softmax update
        float alpha[8];
#pragma unroll
        for (int e = 0; e < 8; ++e) {
            float mn = fmaxf(mrow[e], mblk[e]);
            alpha[e] = __expf(mrow[e] - mn);
            mrow[e] = mn;
            p0e[e] = msk0[e] ? __expf(s0[e] - mn) : 0.0f;
            p1e[e] = msk1[e] ? __expf(s1[e] - mn) : 0.0f;
            float rs = p0e[e] + p1e[e];
#pragma unroll
            for (int off = 1; off < 16; off <<= 1)
                rs += __shfl_xor(rs, off, 16);
            lrow[e] = lrow[e] * alpha[e] + rs;
#pragma unroll
            for (int tI = 0; tI < 4; ++tI) accv[tI][e] *= alpha[e];
            // write P (f16) as row-major 16x32 for A-fragment reload
            lds_p[(e + 8 * hi) * 32 + lo16]      = (_Float16)p0e[e];
            lds_p[(e + 8 * hi) * 32 + 16 + lo16] = (_Float16)p1e[e];
        }
        __syncthreads();  // single-wave WG: ordering no-op, kept as fence

        // P*V: A = P (16x32), B = V^T rows = hd
        const _Float16* pr = &lds_p[lo16 * 32];
        v16h pf = ldfrag(pr + 8 * hi, pr + 16 + 8 * hi);
#pragma unroll
        for (int tI = 0; tI < 4; ++tI) {
            const _Float16* vr = &lds_vt[(tI * 16 + lo16) * 32];
            v16h bf = ldfrag(vr + 8 * hi, vr + 16 + 8 * hi);
            accv[tI] = wmma_f16(pf, bf, accv[tI]);
        }
        __syncthreads();
    }

    // write out: a[(b,s), h*64 + hd] f16
#pragma unroll
    for (int tI = 0; tI < 4; ++tI) {
#pragma unroll
        for (int e = 0; e < 8; ++e) {
            int row = b * SEQ + q0 + e + 8 * hi;
            int col = h * HD + tI * 16 + lo16;
            ah[(size_t)row * H_DIM + col] = (_Float16)(accv[tI][e] / lrow[e]);
        }
    }
}

// ---------------------------------------------------------------------------
// LayerNorm over H=512: y = LN(x (+res)) * g + b. Optional f32/f16 outputs.
// ---------------------------------------------------------------------------
__global__ __launch_bounds__(256)
void ln_kernel(const float* __restrict__ in, const float* __restrict__ res,
               const float* __restrict__ g, const float* __restrict__ bta,
               float* __restrict__ outF, _Float16* __restrict__ outH, float eps) {
    __shared__ float red[256];
    int row = blockIdx.x, tid = threadIdx.x;
    size_t base = (size_t)row * H_DIM;
    float a0 = in[base + tid]       + (res ? res[base + tid]       : 0.0f);
    float a1 = in[base + tid + 256] + (res ? res[base + tid + 256] : 0.0f);
    red[tid] = a0 + a1;
    __syncthreads();
    for (int s = 128; s > 0; s >>= 1) {
        if (tid < s) red[tid] += red[tid + s];
        __syncthreads();
    }
    float mean = red[0] * (1.0f / H_DIM);
    __syncthreads();
    red[tid] = a0 * a0 + a1 * a1;
    __syncthreads();
    for (int s = 128; s > 0; s >>= 1) {
        if (tid < s) red[tid] += red[tid + s];
        __syncthreads();
    }
    float var = red[0] * (1.0f / H_DIM) - mean * mean;
    float rs = rsqrtf(var + eps);
    float y0 = (a0 - mean) * rs * g[tid]       + bta[tid];
    float y1 = (a1 - mean) * rs * g[tid + 256] + bta[tid + 256];
    if (outF) { outF[base + tid] = y0; outF[base + tid + 256] = y1; }
    if (outH) { outH[base + tid] = (_Float16)y0; outH[base + tid + 256] = (_Float16)y1; }
}

// Mean over each patch's Lp tokens
__global__ __launch_bounds__(256)
void pool_kernel(const float* __restrict__ x2, float* __restrict__ qp) {
    int bp = blockIdx.x;
    for (int c = threadIdx.x; c < H_DIM; c += 256) {
        float s = 0.0f;
        for (int t = 0; t < LPATCH; ++t)
            s += x2[((size_t)bp * LPATCH + t) * H_DIM + c];
        qp[(size_t)bp * H_DIM + c] = s * (1.0f / LPATCH);
    }
}

// Small dense: out[row, c] = dot(A[row,:], W[c,:]) + bias[c] (32 rows)
__global__ __launch_bounds__(128)
void smallmm_kernel(const float* __restrict__ A, const float* __restrict__ Wt,
                    const float* __restrict__ bias, float* __restrict__ out,
                    int N, int K) {
    int row = blockIdx.x;
    for (int c = threadIdx.x; c < N; c += 128) {
        const float* a = A + (size_t)row * K;
        const float* w = Wt + (size_t)c * K;
        float s = bias[c];
        for (int k = 0; k < K; ++k) s += a[k] * w[k];
        out[(size_t)row * N + c] = s;
    }
}

// Cross-attention: 1 query x Lp=256 keys per (bp, head). Block = 64 threads.
__global__ __launch_bounds__(64)
void cross_attn_kernel(const float* __restrict__ cq, const float* __restrict__ ckv,
                       float* __restrict__ co) {
    __shared__ float sq[HD];
    __shared__ float sp[LPATCH];
    __shared__ float red[64];
    int bp = blockIdx.x / NHEAD;
    int h  = blockIdx.x % NHEAD;
    int tid = threadIdx.x;
    sq[tid] = cq[(size_t)bp * H_DIM + h * HD + tid];
    __syncthreads();
    for (int k = tid; k < LPATCH; k += 64) {
        const float* kr = ckv + ((size_t)bp * LPATCH + k) * (3 * H_DIM) + H_DIM + h * HD;
        float s = 0.0f;
        for (int d = 0; d < HD; ++d) s += sq[d] * kr[d];
        sp[k] = s * 0.125f;
    }
    __syncthreads();
    float lm = -1e30f;
    for (int k = tid; k < LPATCH; k += 64) lm = fmaxf(lm, sp[k]);
    red[tid] = lm;
    __syncthreads();
    if (tid == 0) {
        float m = red[0];
        for (int i = 1; i < 64; ++i) m = fmaxf(m, red[i]);
        red[0] = m;
    }
    __syncthreads();
    float mx = red[0];
    __syncthreads();
    float ls = 0.0f;
    for (int k = tid; k < LPATCH; k += 64) {
        float e = __expf(sp[k] - mx);
        sp[k] = e;
        ls += e;
    }
    red[tid] = ls;
    __syncthreads();
    if (tid == 0) {
        float s = 0.0f;
        for (int i = 0; i < 64; ++i) s += red[i];
        red[0] = s;
    }
    __syncthreads();
    float denom = red[0];
    float acc = 0.0f;
    for (int k = 0; k < LPATCH; ++k)
        acc += sp[k] * ckv[((size_t)bp * LPATCH + k) * (3 * H_DIM) + 2 * H_DIM + h * HD + tid];
    co[(size_t)bp * H_DIM + h * HD + tid] = acc / denom;
}

// ---------------------------------------------------------------------------
// Host-side orchestration
// ---------------------------------------------------------------------------
extern "C" void kernel_launch(void* const* d_in, const int* in_sizes, int n_in,
                              void* d_out, int out_size, void* d_ws, size_t ws_size,
                              hipStream_t stream) {
    (void)in_sizes; (void)n_in; (void)out_size; (void)ws_size;
    const int*   byte_seq = (const int*)  d_in[0];
    // d_in[1] = n_patches (fixed 8 per reference setup)
    const float* byte_emb = (const float*)d_in[2];
    const float* ng3      = (const float*)d_in[3];
    const float* ng4      = (const float*)d_in[4];
    const float* ng5      = (const float*)d_in[5];
    const float* in_w     = (const float*)d_in[6];
    const float* in_b     = (const float*)d_in[7];
    const float* ao_w     = (const float*)d_in[8];
    const float* ao_b     = (const float*)d_in[9];
    const float* ln1_g    = (const float*)d_in[10];
    const float* ln1_b    = (const float*)d_in[11];
    const float* ff1_w    = (const float*)d_in[12];
    const float* ff1_b    = (const float*)d_in[13];
    const float* ff2_w    = (const float*)d_in[14];
    const float* ff2_b    = (const float*)d_in[15];
    const float* ln2_g    = (const float*)d_in[16];
    const float* ln2_b    = (const float*)d_in[17];
    const float* ca_qkv_w = (const float*)d_in[18];
    const float* ca_qkv_b = (const float*)d_in[19];
    const float* ca_out_w = (const float*)d_in[20];
    const float* ca_out_b = (const float*)d_in[21];
    const float* ca_ln_g  = (const float*)d_in[22];
    const float* ca_ln_b  = (const float*)d_in[23];
    const float* out_ln_g = (const float*)d_in[24];
    const float* out_ln_b = (const float*)d_in[25];

    char* W = (char*)d_ws;
    const size_t MB = 1 << 20;
    _Float16* xh   = (_Float16*)(W + 0);          //  8 MB  (8192 x 512 f16)
    float*    x    = (float*)   (W + 8 * MB);     // 16 MB  (8192 x 512 f32)
    _Float16* qkvh = (_Float16*)(W + 24 * MB);    // 24 MB  (8192 x 1536 f16)
    _Float16* ah   = (_Float16*)(W + 48 * MB);    //  8 MB
    float*    t    = (float*)   (W + 56 * MB);    // 16 MB  (AO out, later FF2 out)
    float*    x1   = (float*)   (W + 72 * MB);    // 16 MB
    _Float16* x1h  = (_Float16*)(W + 88 * MB);    //  8 MB
    _Float16* fh   = (_Float16*)(W + 96 * MB);    // 32 MB  (8192 x 2048 f16)
    float*    x2   = (float*)   (W + 128 * MB);   // 16 MB
    _Float16* knh  = (_Float16*)(W + 144 * MB);   //  8 MB
    float*    ckv  = (float*)   (W + 24 * MB);    // 48 MB  (overlaps dead qkvh/ah/t)
    float*    qpool   = (float*)(W + 152 * MB);   // 32 x 512
    float*    qn      = qpool + 32 * H_DIM;
    float*    cq      = qn    + 32 * H_DIM;
    float*    co      = cq    + 32 * H_DIM;
    float*    patches = co    + 32 * H_DIM;
    _Float16* in_wh   = (_Float16*)(W + 153 * MB);
    _Float16* ao_wh   = in_wh   + 1536 * 512;
    _Float16* ff1h    = ao_wh   + 512 * 512;
    _Float16* ff2h    = ff1h    + 2048 * 512;
    _Float16* ca_qkvh = ff2h    + 512 * 2048;

    // Weight casts to f16
    auto castn = [&](const float* s, _Float16* d, int n) {
        cast_h_kernel<<<(n + 255) / 256, 256, 0, stream>>>(s, d, n);
    };
    castn(in_w,     in_wh,   1536 * 512);
    castn(ao_w,     ao_wh,   512 * 512);
    castn(ff1_w,    ff1h,    2048 * 512);
    castn(ff2_w,    ff2h,    512 * 2048);
    castn(ca_qkv_w, ca_qkvh, 1536 * 512);

    // 1) embedding + n-gram hash gathers -> x (f32), xh (f16)
    embed_kernel<<<ROWS, 128, 0, stream>>>(byte_seq, byte_emb, ng3, ng4, ng5, x, xh);

    // 2) QKV projection: (8192x512) x (1536x512)^T -> qkvh (f16)
    gemm_wmma_kernel<<<dim3(1536 / 64, ROWS / 128), 256, 0, stream>>>(
        xh, in_wh, in_b, nullptr, qkvh, ROWS, 1536, 512, 0);

    // 3) windowed causal attention -> ah (f16)
    attn_kernel<<<BATCH * NHEAD * (SEQ / 16), 32, 0, stream>>>(qkvh, ah);

    // 4) attention output projection -> t (f32)
    gemm_wmma_kernel<<<dim3(512 / 64, ROWS / 128), 256, 0, stream>>>(
        ah, ao_wh, ao_b, t, nullptr, ROWS, 512, 512, 0);

    // 5) x1 = LN(x + t)
    ln_kernel<<<ROWS, 256, 0, stream>>>(x, t, ln1_g, ln1_b, x1, x1h, 1e-5f);

    // 6) FF1 with ReLU -> fh (f16)
    gemm_wmma_kernel<<<dim3(2048 / 64, ROWS / 128), 256, 0, stream>>>(
        x1h, ff1h, ff1_b, nullptr, fh, ROWS, 2048, 512, 1);

    // 7) FF2 -> t (f32)
    gemm_wmma_kernel<<<dim3(512 / 64, ROWS / 128), 256, 0, stream>>>(
        fh, ff2h, ff2_b, t, nullptr, ROWS, 512, 2048, 0);

    // 8) x2 = LN(x1 + t)
    ln_kernel<<<ROWS, 256, 0, stream>>>(x1, t, ln2_g, ln2_b, x2, nullptr, 1e-5f);

    // 9) kn = ca_LN(x2) -> knh (f16 only)
    ln_kernel<<<ROWS, 256, 0, stream>>>(x2, nullptr, ca_ln_g, ca_ln_b,
                                        nullptr, knh, 1e-6f);

    // 10) ckv = kn @ ca_qkv_w^T + b (f32), full 3H
    gemm_wmma_kernel<<<dim3(1536 / 64, ROWS / 128), 256, 0, stream>>>(
        knh, ca_qkvh, ca_qkv_b, ckv, nullptr, ROWS, 1536, 512, 0);

    // 11) qpool = per-patch mean of x2; qn = ca_LN(qpool)
    pool_kernel<<<BATCH * NPATCH, 256, 0, stream>>>(x2, qpool);
    ln_kernel<<<BATCH * NPATCH, 256, 0, stream>>>(qpool, nullptr, ca_ln_g, ca_ln_b,
                                                  qn, nullptr, 1e-6f);

    // 12) cq = qn @ ca_qkv_w[:512]^T + b[:512]
    smallmm_kernel<<<BATCH * NPATCH, 128, 0, stream>>>(qn, ca_qkv_w, ca_qkv_b, cq,
                                                       512, 512);

    // 13) cross attention (1 query x 256 keys per bp,head) -> co
    cross_attn_kernel<<<BATCH * NPATCH * NHEAD, 64, 0, stream>>>(cq, ckv, co);

    // 14) patches = co @ ca_out_w^T + ca_out_b
    smallmm_kernel<<<BATCH * NPATCH, 128, 0, stream>>>(co, ca_out_w, ca_out_b,
                                                       patches, 512, 512);

    // 15) final LN -> d_out (B, P, H) f32
    ln_kernel<<<BATCH * NPATCH, 256, 0, stream>>>(patches, nullptr, out_ln_g, out_ln_b,
                                                  (float*)d_out, nullptr, 1e-6f);
}
